// MambaBlock_25477746000171
// MI455X (gfx1250) — compile-verified, matrix operand order checked by reading, not run on hardware
//
#include <hip/hip_runtime.h>
#include <hip/hip_bf16.h>
#include <stdint.h>

// ---------- feature detection ----------
#ifndef HAVE_TDM
#if defined(__has_builtin)
#if __has_builtin(__builtin_amdgcn_tensor_load_to_lds)
#define HAVE_TDM 1
#endif
#endif
#endif
#ifndef HAVE_TDM
#define HAVE_TDM 0
#endif

#if __has_include(<hip/amd_detail/amd_gfx1250_TDM.h>)
#define TDM_6ARG 1
#else
#define TDM_6ARG 0
#endif

// ---------- types ----------
typedef __bf16 bf16;
typedef __attribute__((ext_vector_type(8)))  __bf16 v8bf;
typedef __attribute__((ext_vector_type(16))) __bf16 v16bf;
typedef __attribute__((ext_vector_type(8)))  float  v8f;
typedef __attribute__((ext_vector_type(4)))  float  v4f;
typedef __attribute__((ext_vector_type(4)))  __bf16 v4bf;
typedef __attribute__((ext_vector_type(4)))  unsigned int v4u;
typedef __attribute__((ext_vector_type(8)))  int v8i;
typedef __attribute__((ext_vector_type(4)))  int v4i;

union Frag16 {
    v16bf v;
    v8bf  h[2];
};

// GEMM tiling
#define BLK_M 128
#define BLK_N 256
#define KS    32          // K-slice per LDS stage
#define LDS_STRIDE 40     // 32 bf16 payload + 8 bf16 pad (16B) -> conflict-free b128 reads

// ---------- fp32 -> bf16 converter (vectorized x4) ----------
__global__ __launch_bounds__(256) void cvt_f32_bf16(const float* __restrict__ in,
                                                    bf16* __restrict__ out, int n) {
    int i = (blockIdx.x * blockDim.x + threadIdx.x) * 4;
    if (i + 3 < n) {
        v4f f = *(const v4f*)(in + i);
        v4bf o;
        o.x = (bf16)f.x; o.y = (bf16)f.y; o.z = (bf16)f.z; o.w = (bf16)f.w;
        *(v4bf*)(out + i) = o;
    }
}

#if HAVE_TDM
// ---------- TDM: DMA a [rows x 32] bf16 tile (row stride ldk elems) into LDS ----------
// LDS destination gets 16B of padding every 64B -> row stride LDS_STRIDE bf16.
// D# built per CDNA5 ISA 8.3-8.6: group0 = {flags, lds_addr, global_addr, type},
// group1 = {data_size/pad, tensor dims, tile dims, dim0 stride}.
__device__ __forceinline__ void tdm_load_tile(uint32_t lds_byte, const bf16* gptr,
                                              uint32_t rows, uint32_t ldk) {
    uint64_t ga = (uint64_t)(uintptr_t)gptr;
    v4u g0;
    g0[0] = 1u;                                           // count=1, user desc
    g0[1] = lds_byte;                                     // lds_addr
    g0[2] = (uint32_t)ga;                                 // global_addr[31:0]
    g0[3] = (uint32_t)((ga >> 32) & 0x1FFFFFFu) | (2u << 30); // addr[56:32] | type=2
    uint32_t td0 = ldk;                                   // tensor dim0 (elements)
    uint32_t td1 = rows;                                  // tensor dim1
    v8i g1;
    g1[0] = (int)((1u << 16)      // data_size = 2 bytes
                | (1u << 20)      // pad_enable
                | (3u << 22)      // pad_interval: 16 DWORDs (=64B)
                | (3u << 25));    // pad_amount:   4 DWORDs (=16B)
    g1[1] = (int)((td0 & 0xFFFFu) << 16);                 // tensor_dim0[15:0]
    g1[2] = (int)((td0 >> 16) | ((td1 & 0xFFFFu) << 16)); // td0[31:16] | td1[15:0]
    g1[3] = (int)((td1 >> 16) | ((uint32_t)KS << 16));    // td1[31:16] | tile_dim0
    g1[4] = (int)rows;                                    // tile_dim1 (tile_dim2=0)
    g1[5] = (int)ldk;                                     // tensor_dim0_stride[31:0]
    g1[6] = 0;                                            // stride hi | dim1_stride lo
    g1[7] = 0;
    v4i g2 = {0, 0, 0, 0};
    v4i g3 = {0, 0, 0, 0};
#if TDM_6ARG
    v8i g4 = {0, 0, 0, 0, 0, 0, 0, 0};
    __builtin_amdgcn_tensor_load_to_lds(g0, g1, g2, g3, g4, 0);
#else
    __builtin_amdgcn_tensor_load_to_lds(g0, g1, g2, g3, 0);
#endif
}
#else
// ---------- fallback: cooperative vectorized copy with the same padded layout ----------
__device__ __forceinline__ void coop_load_tile(bf16* lds, const bf16* gptr,
                                               int rows, int ldk) {
    for (int r = threadIdx.x; r < rows * 4; r += 256) {   // 4 x v8bf chunks per row
        int row = r >> 2, c = (r & 3) * 8;
        *(v8bf*)(lds + row * LDS_STRIDE + c) = *(const v8bf*)(gptr + (size_t)row * ldk + c);
    }
}
#endif

// ---------- bf16 WMMA GEMM:  C[M,N] = A[M,K] * B[N,K]^T + bias[N] ----------
// Block = 256 threads = 8 waves (wave32), waves 2(M) x 4(N), wave tile 64x64
// = 4x4 fragments of 16x16, K staged through LDS in 32-wide double-buffered
// slices moved by the Tensor Data Mover (TENSORcnt) and consumed via ds_load_b128.
template <typename OutT>
__global__ __launch_bounds__(256) void gemm_wmma_bf16(const bf16* __restrict__ A,
                                                      const bf16* __restrict__ B,
                                                      const float* __restrict__ bias,
                                                      OutT* __restrict__ C,
                                                      int M, int N, int K) {
    __shared__ bf16 Ash[2][BLK_M * LDS_STRIDE];   // 2 x 10 KB
    __shared__ bf16 Bsh[2][BLK_N * LDS_STRIDE];   // 2 x 20 KB

    const int lane   = threadIdx.x & 31;
    const int wave   = threadIdx.x >> 5;
    const int wm     = (wave >> 2) * 64;          // wave M base within block
    const int wn     = (wave & 3) * 64;           // wave N base within block
    const int bm     = blockIdx.x * BLK_M;
    const int bn     = blockIdx.y * BLK_N;
    const int lane15 = lane & 15;
    const int khalf  = lane >> 4;

    const bf16* Ag = A + (size_t)bm * K;
    const bf16* Bg = B + (size_t)bn * K;
    const int NK = K / KS;

    v8f acc[4][4] = {};

#if HAVE_TDM
    if (wave == 0) {
        tdm_load_tile((uint32_t)(uintptr_t)&Ash[0][0], Ag, BLK_M, (uint32_t)K);
        tdm_load_tile((uint32_t)(uintptr_t)&Bsh[0][0], Bg, BLK_N, (uint32_t)K);
        if (NK > 1) {
            tdm_load_tile((uint32_t)(uintptr_t)&Ash[1][0], Ag + KS, BLK_M, (uint32_t)K);
            tdm_load_tile((uint32_t)(uintptr_t)&Bsh[1][0], Bg + KS, BLK_N, (uint32_t)K);
        }
    }
#endif

    for (int ik = 0; ik < NK; ++ik) {
        const int buf = ik & 1;
#if HAVE_TDM
        if (wave == 0) {
            if (ik + 1 < NK) __builtin_amdgcn_s_wait_tensorcnt(2);  // slice ik landed
            else             __builtin_amdgcn_s_wait_tensorcnt(0);
        }
#else
        coop_load_tile(Ash[buf], Ag + ik * KS, BLK_M, K);
        coop_load_tile(Bsh[buf], Bg + ik * KS, BLK_N, K);
#endif
        __syncthreads();

        // fragment loads from LDS (ds_load_b128, conflict-free via 40-elem stride)
        Frag16 a[4], b[4];
        const bf16* Al = Ash[buf];
        const bf16* Bl = Bsh[buf];
#pragma unroll
        for (int i = 0; i < 4; ++i) {
            const bf16* p = Al + (wm + i * 16 + lane15) * LDS_STRIDE + khalf * 8;
            a[i].h[0] = *(const v8bf*)(p);
            a[i].h[1] = *(const v8bf*)(p + 16);
        }
#pragma unroll
        for (int j = 0; j < 4; ++j) {
            const bf16* p = Bl + (wn + j * 16 + lane15) * LDS_STRIDE + khalf * 8;
            b[j].h[0] = *(const v8bf*)(p);
            b[j].h[1] = *(const v8bf*)(p + 16);
        }
#pragma unroll
        for (int i = 0; i < 4; ++i)
#pragma unroll
            for (int j = 0; j < 4; ++j)
                acc[i][j] = __builtin_amdgcn_wmma_f32_16x16x32_bf16(
                    false, a[i].v, false, b[j].v, (short)0, acc[i][j], false, false);

        __syncthreads();   // everyone done reading buf before it is overwritten

#if HAVE_TDM
        if (wave == 0 && ik + 2 < NK) {
            tdm_load_tile((uint32_t)(uintptr_t)&Ash[buf][0], Ag + (ik + 2) * KS, BLK_M, (uint32_t)K);
            tdm_load_tile((uint32_t)(uintptr_t)&Bsh[buf][0], Bg + (ik + 2) * KS, BLK_N, (uint32_t)K);
        }
#endif
    }

    // Epilogue: C/D 16x16 f32 layout: VGPR e -> M = khalf*8 + e, lane15 -> N col.
#pragma unroll
    for (int j = 0; j < 4; ++j) {
        const int   col = bn + wn + j * 16 + lane15;
        const float bj  = bias[col];
#pragma unroll
        for (int i = 0; i < 4; ++i) {
            size_t base = (size_t)(bm + wm + i * 16 + khalf * 8) * N + col;
#pragma unroll
            for (int e = 0; e < 8; ++e)
                C[base + (size_t)e * N] = (OutT)(acc[i][j][e] + bj);
        }
    }
}

// ---------- depthwise conv3 (pad 1) + bias + SiLU + gate ----------
// uv: [M, 2D] bf16 (u = cols 0..D-1, v = cols D..2D-1), y: [M, D] bf16
__global__ __launch_bounds__(256) void conv_gate(const bf16* __restrict__ uv,
                                                 const float* __restrict__ Wc,
                                                 const float* __restrict__ bc,
                                                 bf16* __restrict__ y,
                                                 int L, int D) {
    const int D2  = 2 * D;
    const int dpt = D / 4;                      // 4 channels per thread
    int t = blockIdx.x * blockDim.x + threadIdx.x;
    int m = t / dpt;
    int d = (t - m * dpt) * 4;
    int l = m & (L - 1);                        // L is a power of two

    const bf16* row = uv + (size_t)m * D2;
#pragma unroll
    for (int q = 0; q < 4; ++q) {
        int   dd = d + q;
        float v0 = (float)row[D + dd];
        float vm = (l > 0)     ? (float)row[D + dd - D2] : 0.0f;
        float vp = (l < L - 1) ? (float)row[D + dd + D2] : 0.0f;
        float yv = vm * Wc[dd * 3 + 0] + v0 * Wc[dd * 3 + 1] + vp * Wc[dd * 3 + 2] + bc[dd];
        float s  = yv / (1.0f + __expf(-yv));   // silu
        float g  = s * (float)row[dd];          // * u
        y[(size_t)m * D + dd] = (bf16)g;
    }
}

// ---------- launcher ----------
extern "C" void kernel_launch(void* const* d_in, const int* in_sizes, int n_in,
                              void* d_out, int out_size, void* d_ws, size_t ws_size,
                              hipStream_t stream) {
    const float* x  = (const float*)d_in[0];   // [4,4096,1024]
    const float* Wi = (const float*)d_in[1];   // [2048,1024]
    const float* bi = (const float*)d_in[2];   // [2048]
    const float* Wc = (const float*)d_in[3];   // [1024,1,3]
    const float* bc = (const float*)d_in[4];   // [1024]
    const float* Wo = (const float*)d_in[5];   // [1024,1024]
    const float* bo = (const float*)d_in[6];   // [1024]
    float* out = (float*)d_out;

    const int Bb = 4, L = 4096, D = 1024;
    const int M = Bb * L;                      // 16384

    // workspace layout (bytes): Wib 4MB | Wob 2MB | xb 32MB (reused as yb) | uvb 64MB
    char*  ws  = (char*)d_ws;
    bf16*  Wib = (bf16*)ws;
    bf16*  Wob = (bf16*)(ws + (size_t)2 * D * D * 2);
    bf16*  xb  = (bf16*)(ws + (size_t)2 * D * D * 2 + (size_t)D * D * 2);
    bf16*  uvb = (bf16*)((char*)xb + (size_t)M * D * 2);
    bf16*  yb  = xb;                           // xb dead after GEMM1

    // fp32 -> bf16
    cvt_f32_bf16<<<(M * D) / 4 / 256, 256, 0, stream>>>(x, xb, M * D);
    cvt_f32_bf16<<<(2 * D * D) / 4 / 256, 256, 0, stream>>>(Wi, Wib, 2 * D * D);
    cvt_f32_bf16<<<(D * D) / 4 / 256, 256, 0, stream>>>(Wo, Wob, D * D);

    // GEMM1: uv = x @ Wi^T + bi   (out bf16)
    gemm_wmma_bf16<bf16><<<dim3(M / BLK_M, (2 * D) / BLK_N), 256, 0, stream>>>(
        xb, Wib, bi, uvb, M, 2 * D, D);

    // conv + silu + gate -> y (bf16)
    conv_gate<<<(M * (D / 4)) / 256, 256, 0, stream>>>(uvb, Wc, bc, yb, L, D);

    // GEMM2: out = y @ Wo^T + bo  (out fp32)
    gemm_wmma_bf16<float><<<dim3(M / BLK_M, D / BLK_N), 256, 0, stream>>>(
        yb, Wob, bo, out, M, D, D);
}
